// CliffordDDIDecoder_50818053046333
// MI455X (gfx1250) — matrix operand order repaired and checked by reading, not compiled, not run on hardware
//
#include <hip/hip_runtime.h>
#include <math.h>

// ---------------------------------------------------------------------------
// Types for CDNA5 WMMA (wave32)
// ---------------------------------------------------------------------------
typedef __bf16 bf16_t;
typedef __bf16 v16bf __attribute__((ext_vector_type(16)));  // 32 B: full A/B fragment per lane
typedef __bf16 v8bf  __attribute__((ext_vector_type(8)));   // 16 B
typedef float  v8f   __attribute__((ext_vector_type(8)));

#define B_ROWS 16384
#define D_DIM  512
#define H_DIM  256
#define R_DIM  95
#define KJ     64      // K*8 multivector components per sample
#define R_PAD  96      // 95 padded to 6 x 16 WMMA column tiles

// ---------------------------------------------------------------------------
// Compile-time Cl(3,0) Cayley tables, basis order [1,e1,e2,e3,e12,e13,e23,e123]
// ---------------------------------------------------------------------------
struct Cayley {
    int   p[8][8];
    float s[8][8];
    constexpr Cayley() : p(), s() {
        const int mask[8] = {0, 1, 2, 4, 3, 5, 6, 7};
        int inv[8] = {0, 0, 0, 0, 0, 0, 0, 0};
        for (int i = 0; i < 8; ++i) inv[mask[i]] = i;
        for (int i = 0; i < 8; ++i)
            for (int j = 0; j < 8; ++j) {
                int a = mask[i], b = mask[j];
                int swaps = 0, aa = a >> 1;
                while (aa) {
                    int x = aa & b;
                    swaps += (x & 1) + ((x >> 1) & 1) + ((x >> 2) & 1);
                    aa >>= 1;
                }
                p[i][j] = inv[a ^ b];
                s[i][j] = (swaps & 1) ? -1.0f : 1.0f;
            }
    }
};
constexpr Cayley CAY{};

__device__ __forceinline__ v8f wmma_bf16(v16bf a, v16bf b, v8f c) {
    return __builtin_amdgcn_wmma_f32_16x16x32_bf16(false, a, false, b, (short)0, c, false, false);
}

// Fragment layouts (CDNA5 ISA 7.12.2):
//   A frag (16x32): lane = (row&15) + 16*hi, element e = g*8 + o, where
//                   k = g*16 + hi*8 + o  (hi = (k>>3)&1, g = (k>>4)&1, o = k&7)
//   B frag (32x16): lane = (col&15) + 16*hi, element e = g*8 + o (same k map as A)
// Swizzled storage: frag-major [tile][lane][16 elems] -> 32 contiguous bytes/lane.

// ---------------------------------------------------------------------------
// Kernel 0: swizzle weights to bf16 fragment order; build padded T^T fragments
//   w1s: [kc(16)][nt(16)][lane(32)][e(16)]   (K=512, N=256)
//   w2s: [kc(8)][nt(4)][lane][e]             (K=256, N=64)
//   tts: [kc(2)][nt(6)][lane][e]             (K=64,  N=96 padded from 95)
// ---------------------------------------------------------------------------
#define N_W1 (D_DIM * H_DIM)   // 131072
#define N_W2 (H_DIM * KJ)      // 16384
#define N_TT (KJ * R_PAD)      // 6144
#define N_CVT (2 * N_W1 + 2 * N_W2 + N_TT)  // 301056 = 1176 * 256

__global__ void convert_weights_kernel(const float* __restrict__ Wp1,
                                       const float* __restrict__ Wv1,
                                       const float* __restrict__ Wp2,
                                       const float* __restrict__ Wv2,
                                       const float* __restrict__ T,
                                       bf16_t* __restrict__ wp1, bf16_t* __restrict__ wv1,
                                       bf16_t* __restrict__ wp2, bf16_t* __restrict__ wv2,
                                       bf16_t* __restrict__ tt) {
    int i = blockIdx.x * blockDim.x + threadIdx.x;
    if (i >= N_CVT) return;
    if (i < 2 * N_W1) {                    // W1 fragments (both branches)
        int j = (i < N_W1) ? i : i - N_W1;
        int e = j & 15, lane = (j >> 4) & 31, nt = (j >> 9) & 15, kc = j >> 13;
        int k = kc * 32 + ((e >= 8) ? 16 : 0) + ((lane >= 16) ? 8 : 0) + (e & 7);
        int n = nt * 16 + (lane & 15);
        float v = (i < N_W1) ? Wp1[k * H_DIM + n] : Wv1[k * H_DIM + n];
        ((i < N_W1) ? wp1 : wv1)[j] = (bf16_t)v;
    } else if (i < 2 * N_W1 + 2 * N_W2) {  // W2 fragments (both branches)
        int j = i - 2 * N_W1;
        int second = (j >= N_W2);
        if (second) j -= N_W2;
        int e = j & 15, lane = (j >> 4) & 31, nt = (j >> 9) & 3, kc = j >> 11;
        int k = kc * 32 + ((e >= 8) ? 16 : 0) + ((lane >= 16) ? 8 : 0) + (e & 7);
        int n = nt * 16 + (lane & 15);
        float v = second ? Wv2[k * KJ + n] : Wp2[k * KJ + n];
        (second ? wv2 : wp2)[j] = (bf16_t)v;
    } else {                               // T^T fragments, zero-padded to 96 cols
        int j = i - 2 * N_W1 - 2 * N_W2;
        int e = j & 15, lane = (j >> 4) & 31;
        int f = j >> 9, kc = f / 6, nt = f - kc * 6;
        int c = kc * 32 + ((e >= 8) ? 16 : 0) + ((lane >= 16) ? 8 : 0) + (e & 7);   // K index = k*8+j2
        int r = nt * 16 + (lane & 15);
        tt[j] = (r < R_DIM) ? (bf16_t)T[r * KJ + c] : (bf16_t)0.0f;
    }
}

// ---------------------------------------------------------------------------
// Kernel 1: fused projection chain for one branch.
//   X = h(64x512) @ W1 + b1 ; LayerNorm ; exact GELU ; m = X @ W2 + b2
// Block: 256 threads = 8 waves; 64-row tile; grid.x = B/64.
// All hot-loop addresses are (per-thread base) + (compile-time const offset).
// ---------------------------------------------------------------------------
__global__ void __launch_bounds__(256)
proj_kernel(const float* __restrict__ h,
            const bf16_t* __restrict__ w1s, const float* __restrict__ b1,
            const float* __restrict__ lg,   const float* __restrict__ lb,
            const bf16_t* __restrict__ w2s, const float* __restrict__ b2,
            float* __restrict__ mout) {
    // A-fragment-layout staging: [chunkRowGroup/kStep][...][lane][e]
    __shared__ __align__(32) bf16_t As[4][4][32][16];   // 16 KB: 64 rows x 128 K chunk
    __shared__ __align__(32) bf16_t Xs[4][8][32][16];   // 32 KB: 64 rows x 256 cols
    __shared__ float rsum[64][2];
    __shared__ float rsq[64][2];

    const int tid  = threadIdx.x;
    const int lane = tid & 31;
    const int wid  = tid >> 5;
    const int wm   = wid & 3;               // 4 row groups x 16 rows
    const int wn   = wid >> 2;              // 2 col halves x 128 cols
    const int row0 = blockIdx.x * 64;
    const int hi8  = (lane >= 16) ? 8 : 0;  // C/D fragment row offset
    const int c15  = lane & 15;

    // ---- invariant base pointers ------------------------------------------
    // A staging: task = tid + it*256 -> row = (tid>>4) + it*16, kg = tid&15
    const float4* hp0 = (const float4*)(h + (size_t)(row0 + (tid >> 4)) * D_DIM + (tid & 15) * 8);
    bf16_t* as_st = &As[0][(tid & 15) >> 2][(tid >> 4) + 16 * (tid & 1)][((tid >> 1) & 1) * 8];
    const bf16_t* as_ld = &As[wm][0][lane][0];
    const bf16_t* w1w   = w1s + wn * (8 * 512) + lane * 16;   // + (kcg*16+n)*512

    // ---------------- GEMM1: 64x256 = h(64x512) @ W1, K chunks of 128 -------
    v8f acc[8];
#pragma unroll
    for (int n = 0; n < 8; ++n) acc[n] = (v8f)0.0f;

    for (int ck = 0; ck < 4; ++ck) {
        // stage A chunk: 8 consecutive f32 -> one v8bf store per task
#pragma unroll
        for (int it = 0; it < 4; ++it) {
            float4 f0 = hp0[it * 2048 + ck * 32];
            float4 f1 = hp0[it * 2048 + ck * 32 + 1];
            v8bf r;
            r[0] = (bf16_t)f0.x; r[1] = (bf16_t)f0.y; r[2] = (bf16_t)f0.z; r[3] = (bf16_t)f0.w;
            r[4] = (bf16_t)f1.x; r[5] = (bf16_t)f1.y; r[6] = (bf16_t)f1.z; r[7] = (bf16_t)f1.w;
            *(v8bf*)(as_st + it * 2048) = r;                  // As[it] stride = 2048 elems
        }
        __syncthreads();
        const bf16_t* wck = w1w + ck * (4 * 16 * 512);        // this chunk's fragments
#pragma unroll
        for (int kc = 0; kc < 4; ++kc) {
            v16bf af = *(const v16bf*)(as_ld + kc * 512);
#pragma unroll
            for (int n = 0; n < 8; ++n) {
                const v16bf* bp = (const v16bf*)(wck + (kc * 16 + n) * 512);
                acc[n] = wmma_bf16(af, *bp, acc[n]);
            }
        }
        __syncthreads();
    }

    // bias b1 (per-column)
#pragma unroll
    for (int n = 0; n < 8; ++n) {
        float bv = b1[wn * 128 + n * 16 + c15];
#pragma unroll
        for (int v = 0; v < 8; ++v) acc[n][v] += bv;
    }

    // ---------------- LayerNorm stats over 256 cols per row -----------------
#pragma unroll
    for (int v = 0; v < 8; ++v) {
        float s = 0.0f, q = 0.0f;
#pragma unroll
        for (int n = 0; n < 8; ++n) { float x = acc[n][v]; s += x; q += x * x; }
#pragma unroll
        for (int m = 1; m < 16; m <<= 1) {
            s += __shfl_xor(s, m);
            q += __shfl_xor(q, m);
        }
        if (c15 == 0) {
            int row = wm * 16 + v + hi8;
            rsum[row][wn] = s;
            rsq[row][wn]  = q;
        }
    }
    __syncthreads();

    // ---------------- normalize + exact GELU -> Xs in A-fragment layout -----
    // col = wn*128 + n*16 + c15 decomposes: kcIdx = wn*4 + (n>>1),
    // laneIdx = (row&15) + 16*((lane>>3)&1), e = (n&1)*8 + (lane&7)
    float lgv[8], lbv[8];
#pragma unroll
    for (int n = 0; n < 8; ++n) {
        lgv[n] = lg[wn * 128 + n * 16 + c15];
        lbv[n] = lb[wn * 128 + n * 16 + c15];
    }
    bf16_t* xst = &Xs[wm][wn * 4][hi8 + 16 * ((lane >> 3) & 1)][lane & 7];
#pragma unroll
    for (int v = 0; v < 8; ++v) {
        int row = wm * 16 + v + hi8;
        float mu   = (rsum[row][0] + rsum[row][1]) * (1.0f / 256.0f);
        float var  = (rsq[row][0] + rsq[row][1]) * (1.0f / 256.0f) - mu * mu;
        float rstd = rsqrtf(var + 1e-5f);
#pragma unroll
        for (int n = 0; n < 8; ++n) {
            float xn  = (acc[n][v] - mu) * rstd * lgv[n] + lbv[n];
            float gel = 0.5f * xn * (1.0f + erff(xn * 0.70710678118654752f));
            xst[(n >> 1) * 512 + v * 16 + (n & 1) * 8] = (bf16_t)gel;
        }
    }
    __syncthreads();

    // ---------------- GEMM2: 64x64 = X(64x256) @ W2(256x64) -----------------
    const int tm  = wid & 3;
    const int tn0 = (wid >> 2) * 2;
    const bf16_t* xs_ld = &Xs[tm][0][lane][0];
    const bf16_t* w2w   = w2s + tn0 * 512 + lane * 16;        // + (kc*4)*512 + t*512
    v8f acc2[2];
    acc2[0] = (v8f)0.0f; acc2[1] = (v8f)0.0f;
#pragma unroll
    for (int kc = 0; kc < 8; ++kc) {
        v16bf af = *(const v16bf*)(xs_ld + kc * 512);
#pragma unroll
        for (int t = 0; t < 2; ++t) {
            const v16bf* bp = (const v16bf*)(w2w + (kc * 4 + t) * 512);
            acc2[t] = wmma_bf16(af, *bp, acc2[t]);
        }
    }

    // bias b2 + store m (f32): base + compile-time offsets
    float* mo = mout + (size_t)(row0 + tm * 16 + hi8) * KJ + tn0 * 16 + c15;
#pragma unroll
    for (int t = 0; t < 2; ++t) {
        float bv = b2[tn0 * 16 + t * 16 + c15];
#pragma unroll
        for (int v = 0; v < 8; ++v)
            mo[v * KJ + t * 16] = acc2[t][v] + bv;
    }
}

// ---------------------------------------------------------------------------
// Kernel 2: Cayley contraction -> Q (bf16 fragments), out = (Q @ T^T) / 8
// Block: 256 threads = 8 waves; 128-row tile; grid.x = B/128.
// ---------------------------------------------------------------------------
__global__ void __launch_bounds__(256)
combine_kernel(const float* __restrict__ mp, const float* __restrict__ mv,
               const bf16_t* __restrict__ tts, const float* __restrict__ gw,
               float* __restrict__ out) {
    __shared__ __align__(32) bf16_t Qs[8][2][32][16];   // 16 KB: 128 rows x 64 comps
    __shared__ float gws[8];

    const int tid  = threadIdx.x;
    const int lane = tid & 31;
    const int wid  = tid >> 5;
    const int row0 = blockIdx.x * 128;
    const int hi8  = (lane >= 16) ? 8 : 0;
    const int c15  = lane & 15;

    if (tid < 8) gws[tid] = gw[tid];
    __syncthreads();

    // ----- Q[b,k,j2] = sum_i a_i * s(i,j2) * Gv[p(i,j2)],
    //       Gv[m]     = sum_j v_j * s(m,j) * gw[p(m,j)]
    // task = tid + it*256 -> row = (tid>>3) + it*32, k = tid&7 (invariant)
    const float4* ap0 = (const float4*)(mp + (size_t)(row0 + (tid >> 3)) * KJ + (tid & 7) * 8);
    const float4* vp0 = (const float4*)(mv + (size_t)(row0 + (tid >> 3)) * KJ + (tid & 7) * 8);
    // Qs flat = rg*1024 + kc*512 + laneIdx*16 + e ; rg = (tid>>7) + it*2
    bf16_t* qst = &Qs[0][0][0][0] + (tid >> 7) * 1024 + ((tid >> 2) & 1) * 512
                + (((tid >> 3) & 15) + 16 * (tid & 1)) * 16 + ((tid >> 1) & 1) * 8;
#pragma unroll
    for (int it = 0; it < 4; ++it) {
        float4 a0 = ap0[it * 512], a1 = ap0[it * 512 + 1];
        float4 v0 = vp0[it * 512], v1 = vp0[it * 512 + 1];
        float a[8]  = {a0.x, a0.y, a0.z, a0.w, a1.x, a1.y, a1.z, a1.w};
        float vv[8] = {v0.x, v0.y, v0.z, v0.w, v1.x, v1.y, v1.z, v1.w};
        float Gv[8];
#pragma unroll
        for (int m = 0; m < 8; ++m) {
            float s = 0.0f;
#pragma unroll
            for (int j = 0; j < 8; ++j)
                s += CAY.s[m][j] * gws[CAY.p[m][j]] * vv[j];
            Gv[m] = s;
        }
        v8bf qb;
#pragma unroll
        for (int j2 = 0; j2 < 8; ++j2) {
            float s = 0.0f;
#pragma unroll
            for (int i = 0; i < 8; ++i)
                s += CAY.s[i][j2] * a[i] * Gv[CAY.p[i][j2]];
            qb[j2] = (bf16_t)s;
        }
        *(v8bf*)(qst + it * 2048) = qb;
    }
    __syncthreads();

    // ----- readout GEMM: 128x96 = Q(128x64) @ T^T(64x96), scaled by 1/8 -----
    const int tm = wid;                     // each wave: 16 rows x 96 cols
    const bf16_t* qs_ld = &Qs[tm][0][lane][0];
    const bf16_t* ttw   = tts + lane * 16;
    v8f acc[6];
#pragma unroll
    for (int n = 0; n < 6; ++n) acc[n] = (v8f)0.0f;

#pragma unroll
    for (int kc = 0; kc < 2; ++kc) {
        v16bf af = *(const v16bf*)(qs_ld + kc * 512);
#pragma unroll
        for (int n = 0; n < 6; ++n) {
            const v16bf* bp = (const v16bf*)(ttw + (kc * 6 + n) * 512);
            acc[n] = wmma_bf16(af, *bp, acc[n]);
        }
    }

    float* ob = out + (size_t)(row0 + tm * 16 + hi8) * R_DIM;
#pragma unroll
    for (int n = 0; n < 6; ++n) {
        int r = n * 16 + c15;
        if (r < R_DIM) {
#pragma unroll
            for (int v = 0; v < 8; ++v)
                ob[v * R_DIM + r] = acc[n][v] * 0.125f;
        }
    }
}

// ---------------------------------------------------------------------------
// Host launcher
// ---------------------------------------------------------------------------
extern "C" void kernel_launch(void* const* d_in, const int* in_sizes, int n_in,
                              void* d_out, int out_size, void* d_ws, size_t ws_size,
                              hipStream_t stream) {
    const float* h_perp = (const float*)d_in[0];
    const float* h_vuln = (const float*)d_in[1];
    const float* Wp1 = (const float*)d_in[2];
    const float* bp1 = (const float*)d_in[3];
    const float* lgp = (const float*)d_in[4];
    const float* lbp = (const float*)d_in[5];
    const float* Wp2 = (const float*)d_in[6];
    const float* bp2 = (const float*)d_in[7];
    const float* Wv1 = (const float*)d_in[8];
    const float* bv1 = (const float*)d_in[9];
    const float* lgv = (const float*)d_in[10];
    const float* lbv = (const float*)d_in[11];
    const float* Wv2 = (const float*)d_in[12];
    const float* bv2 = (const float*)d_in[13];
    const float* T   = (const float*)d_in[14];
    const float* gw  = (const float*)d_in[15];
    float* out = (float*)d_out;

    // workspace carve-up (bytes, all 4 KB aligned)
    char* ws = (char*)d_ws;
    bf16_t* wp1 = (bf16_t*)(ws);                          // 131072 * 2 = 262144
    bf16_t* wv1 = (bf16_t*)(ws + 262144);                 // 262144
    bf16_t* wp2 = (bf16_t*)(ws + 524288);                 //  32768
    bf16_t* wv2 = (bf16_t*)(ws + 557056);                 //  32768
    bf16_t* tts = (bf16_t*)(ws + 589824);                 //  12288
    float*  m_perp = (float*)(ws + 602112);               // 16384*64*4 = 4194304
    float*  m_vuln = (float*)(ws + 602112 + 4194304);     // 4194304
    // total: 8,990,720 bytes

    convert_weights_kernel<<<N_CVT / 256, 256, 0, stream>>>(
        Wp1, Wv1, Wp2, Wv2, T, wp1, wv1, wp2, wv2, tts);

    dim3 pgrid(B_ROWS / 64);
    proj_kernel<<<pgrid, 256, 0, stream>>>(h_perp, wp1, bp1, lgp, lbp, wp2, bp2, m_perp);
    proj_kernel<<<pgrid, 256, 0, stream>>>(h_vuln, wv1, bv1, lgv, lbv, wv2, bv2, m_vuln);

    combine_kernel<<<B_ROWS / 128, 256, 0, stream>>>(m_perp, m_vuln, tts, gw, out);
}
